// QRNNLayer_84636625535204
// MI455X (gfx1250) — matrix operand order repaired and checked by reading, not compiled, not run on hardware
//
#include <hip/hip_runtime.h>

#define B_   8
#define S_   4096
#define I_   256
#define K_   512
#define N_   1536
#define M_   (B_ * S_)   // 32768
#define G_   512         // per-gate width (2 dirs * 256)

typedef __attribute__((ext_vector_type(16))) __bf16 v16bf;
typedef __attribute__((ext_vector_type(8)))  float  v8f;

union BFrag {
    v16bf v;
    uint4 q[2];   // 32 bytes = 16 bf16
};

__device__ __forceinline__ float fsigmoid(float x) {
    return __builtin_amdgcn_rcpf(1.0f + __expf(-x));
}
__device__ __forceinline__ float ftanh(float x) {
    return 2.0f * fsigmoid(2.0f * x) - 1.0f;   // tanh(x) = 2*sigmoid(2x)-1
}

// ---------------------------------------------------------------------------
// Pre-pass: split f32 -> bf16 hi + bf16 lo  (x = hi + lo)
// ---------------------------------------------------------------------------
__global__ void split_bf16(const float* __restrict__ in,
                           __bf16* __restrict__ hi, __bf16* __restrict__ lo,
                           int n) {
    int i = blockIdx.x * blockDim.x + threadIdx.x;
    if (i < n) {
        float v = in[i];
        __bf16 h = (__bf16)v;
        hi[i] = h;
        lo[i] = (__bf16)(v - (float)h);
    }
}

__global__ void zero_fill_bf16(__bf16* __restrict__ p, int n) {
    int i = blockIdx.x * blockDim.x + threadIdx.x;
    if (i < n) p[i] = (__bf16)0.0f;
}

// ---------------------------------------------------------------------------
// GEMM: g[m][n] = sum_k win[m][k] * W[n][k] + bias[n], activations fused.
//   win[m] = [x[b][s-1][:], x[b][s][:]]  (implicit window; s==0 rows read a
//   zeroed page so the inner loop has no predication at all).
// Block = 256 threads (8 waves), tile 128M x 128N.
// Wave (wm 0..1, wn 0..3) computes 64M x 32N = 4x2 tiles of 16x16.
// Split-bf16: 3 WMMAs (hi*Hi + hi*Lo + lo*Hi) per 16x16x32 K-chunk.
// All inner-loop addresses are base + compile-time immediate offsets.
// ---------------------------------------------------------------------------
__global__ __launch_bounds__(256)
void qrnn_gemm_wmma(const __bf16* __restrict__ xhi, const __bf16* __restrict__ xlo,
                    const __bf16* __restrict__ whi, const __bf16* __restrict__ wlo,
                    const __bf16* __restrict__ zpage,
                    const float* __restrict__ bias,
                    float* __restrict__ zbuf,   // [S][B][512] time-major
                    float* __restrict__ fbuf,   // [S][B][512] time-major
                    float* __restrict__ out) {  // [B][S][512], holds sigmoid(o)
    const int lane    = threadIdx.x & 31;
    const int wave    = threadIdx.x >> 5;
    const int wm      = wave & 1;             // 2 M-groups of 64 rows
    const int wn      = wave >> 1;            // 4 N-groups of 32 cols
    const int mBlock  = blockIdx.x * 128 + wm * 64;
    const int nBlock  = blockIdx.y * 128 + wn * 32;

    const int colLane = lane & 15;
    const int hi16    = lane >> 4;            // 0 for lanes 0-15, 1 for 16-31
    const int aklo    = hi16 * 8;             // A-fragment K-run base (0 / 8)
    const int bkoff   = hi16 * 8;             // B-fragment K-run base (0 / 8)

    // B base pointers: one address computation for the whole kernel.
    const __bf16* pBh[2];
    const __bf16* pBl[2];
#pragma unroll
    for (int nt = 0; nt < 2; ++nt) {
        int n = nBlock + nt * 16 + colLane;
        size_t off = (size_t)n * K_ + bkoff;
        pBh[nt] = whi + off;
        pBl[nt] = wlo + off;
    }

    // A base pointers per M-subtile, for each K half (window: s-1 then s).
    const __bf16* pAh0[4]; const __bf16* pAl0[4];   // K <  256 -> x[s-1]
    const __bf16* pAh1[4]; const __bf16* pAl1[4];   // K >= 256 -> x[s]
#pragma unroll
    for (int mt = 0; mt < 4; ++mt) {
        int rm = mBlock + mt * 16 + colLane;
        int s  = rm & (S_ - 1);
        int bb = rm >> 12;                    // S_ == 2^12
        size_t base1 = ((size_t)(bb * S_ + s) * I_) + aklo;
        pAh1[mt] = xhi + base1;
        pAl1[mt] = xlo + base1;
        if (s > 0) {
            pAh0[mt] = xhi + base1 - I_;
            pAl0[mt] = xlo + base1 - I_;
        } else {                              // zero-padded window row
            pAh0[mt] = zpage + aklo;
            pAl0[mt] = zpage + aklo;
        }
    }

    v8f acc[4][2] = {};

    // One K-chunk of 32: 24 immediate-offset b128 loads + 24 WMMAs.
    auto kstep = [&](const __bf16* const* pAh, const __bf16* const* pAl,
                     int aoff, int boff) {
        BFrag ah[4], al[4], bh[2], bl[2];
#pragma unroll
        for (int mt = 0; mt < 4; ++mt) {
            ah[mt].q[0] = *(const uint4*)(pAh[mt] + aoff);
            ah[mt].q[1] = *(const uint4*)(pAh[mt] + aoff + 16);
            al[mt].q[0] = *(const uint4*)(pAl[mt] + aoff);
            al[mt].q[1] = *(const uint4*)(pAl[mt] + aoff + 16);
        }
#pragma unroll
        for (int nt = 0; nt < 2; ++nt) {
            bh[nt].q[0] = *(const uint4*)(pBh[nt] + boff);
            bh[nt].q[1] = *(const uint4*)(pBh[nt] + boff + 16);
            bl[nt].q[0] = *(const uint4*)(pBl[nt] + boff);
            bl[nt].q[1] = *(const uint4*)(pBl[nt] + boff + 16);
        }
        // Three passes; dependent WMMAs on one accumulator are 8 apart.
#pragma unroll
        for (int mt = 0; mt < 4; ++mt)
#pragma unroll
            for (int nt = 0; nt < 2; ++nt)
                acc[mt][nt] = __builtin_amdgcn_wmma_f32_16x16x32_bf16(
                    false, ah[mt].v, false, bh[nt].v, (short)0, acc[mt][nt], false, false);
#pragma unroll
        for (int mt = 0; mt < 4; ++mt)
#pragma unroll
            for (int nt = 0; nt < 2; ++nt)
                acc[mt][nt] = __builtin_amdgcn_wmma_f32_16x16x32_bf16(
                    false, ah[mt].v, false, bl[nt].v, (short)0, acc[mt][nt], false, false);
#pragma unroll
        for (int mt = 0; mt < 4; ++mt)
#pragma unroll
            for (int nt = 0; nt < 2; ++nt)
                acc[mt][nt] = __builtin_amdgcn_wmma_f32_16x16x32_bf16(
                    false, al[mt].v, false, bh[nt].v, (short)0, acc[mt][nt], false, false);
    };

#pragma unroll
    for (int kc = 0; kc < 8; ++kc)            // window half: x[s-1]
        kstep(pAh0, pAl0, kc * 32, kc * 32);
#pragma unroll
    for (int kc = 0; kc < 8; ++kc)            // current half: x[s]
        kstep(pAh1, pAl1, kc * 32, I_ + kc * 32);

    // Epilogue. C layout: VGPR r -> (M = r + 8*hi16, N = lane&15).
#pragma unroll
    for (int nt = 0; nt < 2; ++nt) {
        const int n    = nBlock + nt * 16 + colLane;
        const float bv = bias[n];
        const int gate = n >> 9;              // 0:z(tanh) 1:f(sig) 2:o(sig)
        const int idx  = n & 511;             // dir*256 + channel
#pragma unroll
        for (int mt = 0; mt < 4; ++mt) {
            const int mrow0 = mBlock + mt * 16 + hi16 * 8;
            v8f a = acc[mt][nt];
#pragma unroll
            for (int r = 0; r < 8; ++r) {
                int m  = mrow0 + r;
                int bb = m >> 12;
                int s  = m & (S_ - 1);
                float g = a[r] + bv;
                if (gate == 0)
                    zbuf[(size_t)s * (B_ * G_) + bb * G_ + idx] = ftanh(g);
                else if (gate == 1)
                    fbuf[(size_t)s * (B_ * G_) + bb * G_ + idx] = fsigmoid(g);
                else
                    out[((size_t)bb * S_ + s) * G_ + idx] = fsigmoid(g);
            }
        }
    }
}

// ---------------------------------------------------------------------------
// Scan: 4096 independent recurrences c[t] = f*c + (1-f)*z (fwd dir=0, bwd
// dir=1), coalesced time-major reads; out already holds sigmoid(o).
// ---------------------------------------------------------------------------
__global__ __launch_bounds__(256)
void qrnn_scan(const float* __restrict__ zbuf, const float* __restrict__ fbuf,
               float* __restrict__ out) {
    const int tid = blockIdx.x * blockDim.x + threadIdx.x;  // 0..4095
    const int bb  = tid >> 9;
    const int idx = tid & 511;
    const int dir = idx >> 8;
    const size_t obase = (size_t)bb * S_ * G_ + idx;
    float c = 0.0f;
    if (dir == 0) {
#pragma unroll 4
        for (int t = 0; t < S_; ++t) {
            size_t k = (size_t)t * (B_ * G_) + tid;
            float f = fbuf[k], z = zbuf[k];
            c = f * c + (1.0f - f) * z;
            size_t oo = obase + (size_t)t * G_;
            out[oo] = out[oo] * c;
        }
    } else {
#pragma unroll 4
        for (int t = S_ - 1; t >= 0; --t) {
            size_t k = (size_t)t * (B_ * G_) + tid;
            float f = fbuf[k], z = zbuf[k];
            c = f * c + (1.0f - f) * z;
            size_t oo = obase + (size_t)t * G_;
            out[oo] = out[oo] * c;
        }
    }
}

// ---------------------------------------------------------------------------
extern "C" void kernel_launch(void* const* d_in, const int* in_sizes, int n_in,
                              void* d_out, int out_size, void* d_ws, size_t ws_size,
                              hipStream_t stream) {
    const float* x    = (const float*)d_in[0];
    const float* W    = (const float*)d_in[1];
    const float* bias = (const float*)d_in[2];
    float* out = (float*)d_out;

    char* ws = (char*)d_ws;
    size_t off = 0;
    auto take = [&](size_t bytes) -> char* {
        char* p = ws + off;
        off += (bytes + 255) & ~(size_t)255;
        return p;
    };
    const size_t NX = (size_t)B_ * S_ * I_;   // 8,388,608
    const size_t NW = (size_t)N_ * K_;        // 786,432
    const int    NZ = 512;                    // zero page elements (1 KB)

    __bf16* xhi   = (__bf16*)take(NX * 2);
    __bf16* xlo   = (__bf16*)take(NX * 2);
    __bf16* whi   = (__bf16*)take(NW * 2);
    __bf16* wlo   = (__bf16*)take(NW * 2);
    __bf16* zpage = (__bf16*)take(NZ * 2);
    float*  zbuf  = (float*)take((size_t)S_ * B_ * G_ * 4);
    float*  fbuf  = (float*)take((size_t)S_ * B_ * G_ * 4);

    split_bf16<<<(int)((NX + 255) / 256), 256, 0, stream>>>(x, xhi, xlo, (int)NX);
    split_bf16<<<(int)((NW + 255) / 256), 256, 0, stream>>>(W, whi, wlo, (int)NW);
    zero_fill_bf16<<<2, 256, 0, stream>>>(zpage, NZ);

    dim3 grid(M_ / 128, N_ / 128);            // 256 x 12
    qrnn_gemm_wmma<<<grid, 256, 0, stream>>>(xhi, xlo, whi, wlo, zpage, bias,
                                             zbuf, fbuf, out);

    qrnn_scan<<<(B_ * G_) / 256, 256, 0, stream>>>(zbuf, fbuf, out);
}
